// IAGNN_47957604827509
// MI455X (gfx1250) — matrix operand order repaired and checked by reading, not compiled, not run on hardware
//
#include <hip/hip_runtime.h>
#include <hip/hip_bf16.h>

typedef __attribute__((ext_vector_type(16))) __bf16 v16bf;
typedef __attribute__((ext_vector_type(8)))  __bf16 v8bf;
typedef __attribute__((ext_vector_type(8)))  float  v8f;

#define NI_  100000
#define NC_  1024
#define B_   512
#define N_   50
#define L_   50
#define D_   128
#define E_   (NI_ * 3)

__device__ __forceinline__ float wave_sum(float v) {
  #pragma unroll
  for (int m = 16; m >= 1; m >>= 1) v += __shfl_xor(v, m, 32);
  return v;
}
__device__ __forceinline__ float lrelu(float x) { return x >= 0.f ? x : 0.2f * x; }

// ---------------- row L2 normalize (f32 out + bf16 out) ----------------
__global__ void k_row_normalize(const float* __restrict__ in, float* __restrict__ outf,
                                __bf16* __restrict__ outbf, int rows) {
  int wid = threadIdx.x >> 5, lane = threadIdx.x & 31;
  int row = blockIdx.x * (blockDim.x >> 5) + wid;
  if (row >= rows) return;
  const float4 v = *(const float4*)(in + (size_t)row * D_ + lane * 4);
  float ss = v.x * v.x + v.y * v.y + v.z * v.z + v.w * v.w;
  ss = wave_sum(ss);
  float r = rsqrtf(ss > 0.f ? ss : 1.f);
  float4 o = {v.x * r, v.y * r, v.z * r, v.w * r};
  *(float4*)(outf + (size_t)row * D_ + lane * 4) = o;
  __bf16* p = outbf + (size_t)row * D_ + lane * 4;
  p[0] = (__bf16)o.x; p[1] = (__bf16)o.y; p[2] = (__bf16)o.z; p[3] = (__bf16)o.w;
}

// ---------------- 128x128 transpose + bf16 convert ----------------
__global__ void k_transpose_bf(const float* __restrict__ W, __bf16* __restrict__ WT) {
  int t = blockIdx.x * blockDim.x + threadIdx.x;
  if (t >= D_ * D_) return;
  int n = t >> 7, k = t & 127;
  WT[n * D_ + k] = (__bf16)W[k * D_ + n];
}

__global__ void k_zero_u32(unsigned* __restrict__ p, int n) {
  int t = blockIdx.x * blockDim.x + threadIdx.x;
  if (t < n) p[t] = 0u;
}

// ---------------- generic C = A[MxK] @ B[NxK]^T * scale (bf16 WMMA) ----------------
template <bool OUTBF>
__global__ void k_gemm_bt_bf16(const __bf16* __restrict__ A, const __bf16* __restrict__ B,
                               void* __restrict__ Cout, int M, int N, int K, float scale) {
  const int lane = threadIdx.x & 31;
  const int hlf = lane >> 4, l16 = lane & 15;
  const int tm = blockIdx.x * 16;
  const int tn = blockIdx.y * 16;
  int arow = tm + l16; if (arow >= M) arow = M - 1;
  int brow = tn + l16; if (brow >= N) brow = N - 1;
  const __bf16* Ap = A + (size_t)arow * K + hlf * 16;
  const __bf16* Bp = B + (size_t)brow * K + hlf * 16;
  v8f c = {};
  for (int k0 = 0; k0 < K; k0 += 32) {
    union { v16bf v; v8bf h[2]; } ua;
    ua.h[0] = *(const v8bf*)(Ap + k0);
    ua.h[1] = *(const v8bf*)(Ap + k0 + 8);
    v16bf b = *(const v16bf*)(Bp + k0);
    if (k0 + 32 < K) __builtin_prefetch((const void*)(Bp + k0 + 32), 0, 0);
    c = __builtin_amdgcn_wmma_f32_16x16x32_bf16(false, ua.v, false, b, (short)0, c,
                                                false, false);
  }
  #pragma unroll
  for (int v = 0; v < 8; ++v) {
    int row = tm + v + hlf * 8;
    int col = tn + l16;
    if (row < M && col < N) {
      float x = c[v] * scale;
      if (OUTBF) ((__bf16*)Cout)[(size_t)row * N + col] = (__bf16)x;
      else       ((float*)Cout)[(size_t)row * N + col] = x;
    }
  }
}

// ---------------- fused sim (itemP @ intentP^T / sqrt(D)) + top-3 ----------------
__global__ void k_sim_topk(const __bf16* __restrict__ itemP, const __bf16* __restrict__ intentP,
                           int* __restrict__ idx3) {
  __shared__ float sims[16 * NC_];          // 64 KB
  const int lane = threadIdx.x & 31, wid = threadIdx.x >> 5; // 4 waves
  const int hlf = lane >> 4, l16 = lane & 15;
  const int tm = blockIdx.x * 16;
  const __bf16* Ap = itemP + (size_t)(tm + l16) * D_ + hlf * 16;
  v16bf areg[4];
  #pragma unroll
  for (int kb = 0; kb < 4; ++kb) {
    union { v16bf v; v8bf h[2]; } ua;
    ua.h[0] = *(const v8bf*)(Ap + kb * 32);
    ua.h[1] = *(const v8bf*)(Ap + kb * 32 + 8);
    areg[kb] = ua.v;
  }
  const float scale = 0.08838834764831845f; // 1/sqrt(128)
  for (int t = 0; t < 16; ++t) {
    int tn = (wid * 16 + t) * 16;
    const __bf16* Bp = intentP + (size_t)(tn + l16) * D_ + hlf * 16;
    v8f c = {};
    #pragma unroll
    for (int kb = 0; kb < 4; ++kb) {
      v16bf b = *(const v16bf*)(Bp + kb * 32);
      c = __builtin_amdgcn_wmma_f32_16x16x32_bf16(false, areg[kb], false, b, (short)0, c,
                                                  false, false);
    }
    #pragma unroll
    for (int v = 0; v < 8; ++v)
      sims[(v + hlf * 8) * NC_ + tn + l16] = c[v] * scale;
  }
  __syncthreads();
  if (threadIdx.x < 16) {
    const float* row = sims + threadIdx.x * NC_;
    float v0 = -3.4e38f, v1 = -3.4e38f, v2 = -3.4e38f; int i0 = 0, i1 = 0, i2 = 0;
    for (int j = 0; j < NC_; ++j) {
      float x = row[j];
      if (x > v0)      { v2 = v1; i2 = i1; v1 = v0; i1 = i0; v0 = x; i0 = j; }
      else if (x > v1) { v2 = v1; i2 = i1; v1 = x;  i1 = j; }
      else if (x > v2) { v2 = x;  i2 = j; }
    }
    size_t base = (size_t)(tm + threadIdx.x) * 3;
    idx3[base] = i0; idx3[base + 1] = i1; idx3[base + 2] = i2;
  }
}

// ---------------- edge kernels: segment softmax over intents ----------------
__global__ void k_edge_e1(const float* __restrict__ item_w, const float* __restrict__ intent_w,
                          const float* __restrict__ la, const int* __restrict__ idx3,
                          float* __restrict__ e1, unsigned* __restrict__ m1u, int E) {
  int wid = threadIdx.x >> 5, lane = threadIdx.x & 31;
  int e = blockIdx.x * (blockDim.x >> 5) + wid;
  if (e >= E) return;
  int i = e / 3, c = idx3[e];
  const float* iw = item_w + (size_t)i * D_;
  const float* cw = intent_w + (size_t)c * D_;
  float s = 0.f;
  #pragma unroll
  for (int q = 0; q < 4; ++q) { int d = lane * 4 + q; s += iw[d] * cw[d] * la[d]; }
  s = wave_sum(s);
  float v = lrelu(s);
  if (lane == 0) {
    e1[e] = v;
    unsigned u = __float_as_uint(v);
    u = (u & 0x80000000u) ? ~u : (u | 0x80000000u);   // order-preserving encode
    atomicMax(m1u + c, u);
  }
}

__global__ void k_edge_ex(const float* __restrict__ e1, const int* __restrict__ idx3,
                          const unsigned* __restrict__ m1u, float* __restrict__ ex1,
                          float* __restrict__ den1, int E) {
  int e = blockIdx.x * blockDim.x + threadIdx.x;
  if (e >= E) return;
  int c = idx3[e];
  unsigned u = m1u[c];
  float m = (u & 0x80000000u) ? __uint_as_float(u ^ 0x80000000u) : __uint_as_float(~u);
  float x = __expf(e1[e] - m);
  ex1[e] = x;
  atomicAdd(den1 + c, x);
}

__global__ void k_edge_scatter(const float* __restrict__ item_w, const int* __restrict__ idx3,
                               const float* __restrict__ ex1, const float* __restrict__ den1,
                               float* __restrict__ intent_new, int E) {
  int wid = threadIdx.x >> 5, lane = threadIdx.x & 31;
  int e = blockIdx.x * (blockDim.x >> 5) + wid;
  if (e >= E) return;
  int i = e / 3, c = idx3[e];
  float w = ex1[e] / den1[c];
  const float* iw = item_w + (size_t)i * D_;
  float* out = intent_new + (size_t)c * D_;
  #pragma unroll
  for (int q = 0; q < 4; ++q) { int d = lane * 4 + q; atomicAdd(out + d, w * iw[d]); }
}

// ---------------- per-item: softmax over its 3 intents, agg ----------------
__global__ void k_item_nei(const float* __restrict__ item_w, const float* __restrict__ intent_new,
                           const float* __restrict__ lb, const int* __restrict__ idx3,
                           float* __restrict__ agg, int n_items) {
  int wid = threadIdx.x >> 5, lane = threadIdx.x & 31;
  int i = blockIdx.x * (blockDim.x >> 5) + wid;
  if (i >= n_items) return;
  const float* iw  = item_w + (size_t)i * D_;
  const float* nb0 = intent_new + (size_t)idx3[i * 3 + 0] * D_;
  const float* nb1 = intent_new + (size_t)idx3[i * 3 + 1] * D_;
  const float* nb2 = intent_new + (size_t)idx3[i * 3 + 2] * D_;
  float s0 = 0.f, s1 = 0.f, s2 = 0.f;
  #pragma unroll
  for (int q = 0; q < 4; ++q) {
    int d = lane * 4 + q;
    float w = iw[d] * lb[d];
    s0 += w * nb0[d]; s1 += w * nb1[d]; s2 += w * nb2[d];
  }
  s0 = lrelu(wave_sum(s0)); s1 = lrelu(wave_sum(s1)); s2 = lrelu(wave_sum(s2));
  float m = fmaxf(s0, fmaxf(s1, s2));
  float w0 = __expf(s0 - m), w1 = __expf(s1 - m), w2 = __expf(s2 - m);
  float inv = 1.f / (w0 + w1 + w2);
  w0 *= inv; w1 *= inv; w2 *= inv;
  #pragma unroll
  for (int q = 0; q < 4; ++q) {
    int d = lane * 4 + q;
    float nei = w0 * nb0[d] + w1 * nb1[d] + w2 * nb2[d];
    agg[(size_t)i * D_ + d] = 0.5f * iw[d] + 0.5f * nei;
  }
}

// ---------------- hs: masked mean of item_w[inputs] ----------------
__global__ void k_hs(const float* __restrict__ item_w, const int* __restrict__ inputs,
                     const int* __restrict__ masks, float* __restrict__ hs) {
  int wid = threadIdx.x >> 5, lane = threadIdx.x & 31;
  int b = blockIdx.x * (blockDim.x >> 5) + wid;
  if (b >= B_) return;
  float a0 = 0.f, a1 = 0.f, a2 = 0.f, a3 = 0.f, msum = 0.f;
  for (int l = 0; l < L_; ++l) {
    float mk = (float)masks[b * L_ + l];
    msum += mk;
    const float* p = item_w + (size_t)inputs[b * L_ + l] * D_ + lane * 4;
    a0 += mk * p[0]; a1 += mk * p[1]; a2 += mk * p[2]; a3 += mk * p[3];
  }
  float* o = hs + b * D_ + lane * 4;
  o[0] = a0 / msum; o[1] = a1 / msum; o[2] = a2 / msum; o[3] = a3 / msum;
}

// ---------------- per-session 50x50 attention, two edge types ----------------
__global__ void k_local(const float* __restrict__ item_w, const float* __restrict__ agg,
                        const float* __restrict__ hs, const int* __restrict__ items,
                        const int* __restrict__ Amat,
                        const float* __restrict__ la1, const float* __restrict__ la2,
                        const float* __restrict__ lb1, const float* __restrict__ lb2,
                        float* __restrict__ outBND) {
  __shared__ float hi[N_ * D_];
  __shared__ float hin[N_ * D_];
  __shared__ float hsb[D_];
  __shared__ float sc1[N_];
  __shared__ float sc2[N_];
  int b = blockIdx.x, tid = threadIdx.x, lane = tid & 31, wid = tid >> 5;
  for (int t = tid; t < N_ * D_; t += 256) {
    int n = t >> 7, d = t & 127;
    int it = items[b * N_ + n];
    hi[t]  = item_w[(size_t)it * D_ + d];
    hin[t] = agg[(size_t)it * D_ + d];
  }
  if (tid < D_) hsb[tid] = hs[b * D_ + tid];
  __syncthreads();
  for (int j = wid; j < N_; j += 8) {
    float s1 = 0.f, s2 = 0.f;
    #pragma unroll
    for (int q = 0; q < 4; ++q) {
      int d = lane * 4 + q;
      float h = hi[j * D_ + d] * hsb[d];
      s1 += h * lb1[d]; s2 += h * lb2[d];
    }
    s1 = wave_sum(s1); s2 = wave_sum(s2);
    if (lane == 0) { sc1[j] = s1; sc2[j] = s2; }
  }
  __syncthreads();
  for (int i = wid; i < N_; i += 8) {
    // pass 1: online softmax stats (uniform across the wave)
    float m = -3.4e38f, s = 0.f;
    for (int j = 0; j < N_; ++j) {
      float d1 = 0.f, d2 = 0.f;
      #pragma unroll
      for (int q = 0; q < 4; ++q) {
        int d = lane * 4 + q;
        float p = hi[i * D_ + d] * hi[j * D_ + d];
        d1 += p * la1[d]; d2 += p * la2[d];
      }
      d1 = wave_sum(d1); d2 = wave_sum(d2);
      int a = Amat[((size_t)b * N_ + i) * N_ + j];
      float att = (a == 1) ? lrelu(d1 + sc1[j])
                : (a == 2) ? lrelu(d2 + sc2[j]) : -9e15f;
      float mn = fmaxf(m, att);
      s = s * __expf(m - mn) + __expf(att - mn);
      m = mn;
    }
    float inv_s = 1.f / s;
    // pass 2: weighted accumulation (recompute attention logits)
    float a0 = 0.f, a1 = 0.f, a2 = 0.f, a3 = 0.f;
    for (int j = 0; j < N_; ++j) {
      float d1 = 0.f, d2 = 0.f;
      #pragma unroll
      for (int q = 0; q < 4; ++q) {
        int d = lane * 4 + q;
        float p = hi[i * D_ + d] * hi[j * D_ + d];
        d1 += p * la1[d]; d2 += p * la2[d];
      }
      d1 = wave_sum(d1); d2 = wave_sum(d2);
      int a = Amat[((size_t)b * N_ + i) * N_ + j];
      float att = (a == 1) ? lrelu(d1 + sc1[j])
                : (a == 2) ? lrelu(d2 + sc2[j]) : -9e15f;
      float p = __expf(att - m) * inv_s;
      int base = j * D_ + lane * 4;
      a0 += p * hi[base]; a1 += p * hi[base + 1]; a2 += p * hi[base + 2]; a3 += p * hi[base + 3];
    }
    size_t ob = ((size_t)b * N_ + i) * D_ + lane * 4;
    int hb = i * D_ + lane * 4;
    float hv;
    hv = hin[hb];     outBND[ob]     = a0 + hv + a0 * hv;
    hv = hin[hb + 1]; outBND[ob + 1] = a1 + hv + a1 * hv;
    hv = hin[hb + 2]; outBND[ob + 2] = a2 + hv + a2 * hv;
    hv = hin[hb + 3]; outBND[ob + 3] = a3 + hv + a3 * hv;
  }
}

// ---------------- readout: alias gather, hir(+pos), hsr ----------------
__global__ void k_sessions(const float* __restrict__ outBND, const int* __restrict__ alias,
                           const int* __restrict__ masks, const float* __restrict__ pos_emb,
                           float* __restrict__ sessions, __bf16* __restrict__ hir_bf,
                           float* __restrict__ hsr, __bf16* __restrict__ hsr_bf) {
  __shared__ float hsum[D_];
  __shared__ float msum;
  int b = blockIdx.x, tid = threadIdx.x;
  if (tid < D_) hsum[tid] = 0.f;
  if (tid == 0) {
    float s = 0.f;
    for (int l = 0; l < L_; ++l) s += (float)masks[b * L_ + l];
    msum = s;
  }
  __syncthreads();
  for (int t = tid; t < L_ * D_; t += 256) {
    int l = t >> 7, d = t & 127;
    int n = alias[b * L_ + l];
    float v = outBND[((size_t)b * N_ + n) * D_ + d];
    sessions[((size_t)b * L_ + l) * D_ + d] = v;
    float hv = v + pos_emb[l * D_ + d];
    hir_bf[((size_t)b * L_ + l) * D_ + d] = (__bf16)hv;
    atomicAdd(&hsum[d], (float)masks[b * L_ + l] * hv);
  }
  __syncthreads();
  if (tid < D_) {
    float v = hsum[tid] / msum;
    hsr[b * D_ + tid] = v;
    hsr_bf[b * D_ + tid] = (__bf16)v;
  }
}

// ---------------- al[b,l] = sigmoid(q1+b1+q2+b2) . w3 ----------------
__global__ void k_al(const float* __restrict__ q1, const float* __restrict__ q2,
                     const float* __restrict__ b1, const float* __restrict__ b2,
                     const float* __restrict__ w3, float* __restrict__ al) {
  int wid = threadIdx.x >> 5, lane = threadIdx.x & 31;
  int bl = blockIdx.x * (blockDim.x >> 5) + wid;
  if (bl >= B_ * L_) return;
  int b = bl / L_;
  float s = 0.f;
  #pragma unroll
  for (int q = 0; q < 4; ++q) {
    int d = lane * 4 + q;
    float z = q1[(size_t)bl * D_ + d] + b1[d] + q2[b * D_ + d] + b2[d];
    s += w3[d] / (1.f + __expf(-z));
  }
  s = wave_sum(s);
  if (lane == 0) al[bl] = s;
}

// ---------------- hf = sum_l al*sessions*mask + pos_n ----------------
__global__ void k_hf(const float* __restrict__ sessions, const float* __restrict__ al,
                     const int* __restrict__ masks, const float* __restrict__ pos_emb,
                     float* __restrict__ hf, __bf16* __restrict__ hf_bf) {
  int b = blockIdx.x, d = threadIdx.x;   // 128 threads
  float acc = 0.f, psum = 0.f, msum = 0.f;
  for (int l = 0; l < L_; ++l) {
    float mk = (float)masks[b * L_ + l];
    msum += mk;
    acc  += al[b * L_ + l] * sessions[((size_t)b * L_ + l) * D_ + d] * mk;
    psum += mk * pos_emb[l * D_ + d];
  }
  float v = acc + pos_emb[L_ * D_ + d] * psum / msum;   // pos_emb[-1] row = MAX_LEN
  hf[b * D_ + d] = v;
  hf_bf[b * D_ + d] = (__bf16)v;
}

extern "C" void kernel_launch(void* const* d_in, const int* in_sizes, int n_in,
                              void* d_out, int out_size, void* d_ws, size_t ws_size,
                              hipStream_t stream) {
  (void)in_sizes; (void)n_in; (void)out_size; (void)ws_size;
  const int* items  = (const int*)d_in[0];
  const int* Amat   = (const int*)d_in[1];
  const int* inputs = (const int*)d_in[2];
  const int* masks  = (const int*)d_in[3];
  const int* alias  = (const int*)d_in[4];
  const float* item_emb   = (const float*)d_in[5];
  const float* intent_emb = (const float*)d_in[6];
  const float* pos_emb    = (const float*)d_in[7];
  const float* lq  = (const float*)d_in[8];
  const float* lk  = (const float*)d_in[9];
  const float* la  = (const float*)d_in[10];
  const float* lb  = (const float*)d_in[11];
  const float* la1 = (const float*)d_in[12];
  const float* la2 = (const float*)d_in[13];
  const float* lb1 = (const float*)d_in[14];
  const float* lb2 = (const float*)d_in[15];
  const float* W1  = (const float*)d_in[16];
  const float* b1  = (const float*)d_in[17];
  const float* W2  = (const float*)d_in[18];
  const float* b2  = (const float*)d_in[19];
  const float* w3  = (const float*)d_in[20];

  char* ws = (char*)d_ws;
  size_t off = 0;
  auto alloc = [&](size_t bytes) -> char* {
    char* p = ws + off;
    off = (off + bytes + 255) & ~(size_t)255;
    return p;
  };
  float*  item_w      = (float*) alloc((size_t)NI_ * D_ * 4);
  __bf16* item_w_bf   = (__bf16*)alloc((size_t)NI_ * D_ * 2);
  float*  intent_w    = (float*) alloc((size_t)NC_ * D_ * 4);
  __bf16* intent_w_bf = (__bf16*)alloc((size_t)NC_ * D_ * 2);
  __bf16* lqT_bf      = (__bf16*)alloc(D_ * D_ * 2);
  __bf16* lkT_bf      = (__bf16*)alloc(D_ * D_ * 2);
  __bf16* W1T_bf      = (__bf16*)alloc(D_ * D_ * 2);
  __bf16* W2T_bf      = (__bf16*)alloc(D_ * D_ * 2);
  __bf16* itemP_bf    = (__bf16*)alloc((size_t)NI_ * D_ * 2);
  __bf16* intentP_bf  = (__bf16*)alloc((size_t)NC_ * D_ * 2);
  int*    idx3        = (int*)   alloc((size_t)E_ * 4);
  float*  e1          = (float*) alloc((size_t)E_ * 4);
  float*  ex1         = (float*) alloc((size_t)E_ * 4);
  unsigned* m1u       = (unsigned*)alloc(NC_ * 4);
  float*  den1        = (float*) alloc(NC_ * 4);
  float*  intent_new  = (float*) alloc((size_t)NC_ * D_ * 4);
  float*  agg         = (float*) alloc((size_t)NI_ * D_ * 4);
  float*  hs          = (float*) alloc((size_t)B_ * D_ * 4);
  float*  outBND      = (float*) alloc((size_t)B_ * N_ * D_ * 4);
  float*  sessions    = (float*) alloc((size_t)B_ * L_ * D_ * 4);
  __bf16* hir_bf      = (__bf16*)alloc((size_t)B_ * L_ * D_ * 2);
  float*  hsr         = (float*) alloc((size_t)B_ * D_ * 4);
  __bf16* hsr_bf      = (__bf16*)alloc((size_t)B_ * D_ * 2);
  float*  q1          = (float*) alloc((size_t)B_ * L_ * D_ * 4);
  float*  q2          = (float*) alloc((size_t)B_ * D_ * 4);
  float*  al          = (float*) alloc((size_t)B_ * L_ * 4);
  float*  hf          = (float*) alloc((size_t)B_ * D_ * 4);
  __bf16* hf_bf       = (__bf16*)alloc((size_t)B_ * D_ * 2);

  // 1) normalize embedding tables
  k_row_normalize<<<(NI_ + 3) / 4, 128, 0, stream>>>(item_emb, item_w, item_w_bf, NI_);
  k_row_normalize<<<(NC_ + 3) / 4, 128, 0, stream>>>(intent_emb, intent_w, intent_w_bf, NC_);
  // 2) transpose weights to [N][K] bf16
  k_transpose_bf<<<64, 256, 0, stream>>>(lq, lqT_bf);
  k_transpose_bf<<<64, 256, 0, stream>>>(lk, lkT_bf);
  k_transpose_bf<<<64, 256, 0, stream>>>(W1, W1T_bf);
  k_transpose_bf<<<64, 256, 0, stream>>>(W2, W2T_bf);
  // 3) projections (WMMA)
  k_gemm_bt_bf16<true><<<dim3(NI_ / 16, D_ / 16), 32, 0, stream>>>(
      item_w_bf, lqT_bf, itemP_bf, NI_, D_, D_, 1.f);
  k_gemm_bt_bf16<true><<<dim3(NC_ / 16, D_ / 16), 32, 0, stream>>>(
      intent_w_bf, lkT_bf, intentP_bf, NC_, D_, D_, 1.f);
  // 4) fused sim GEMM + top-3 routing (WMMA into LDS)
  k_sim_topk<<<NI_ / 16, 128, 0, stream>>>(itemP_bf, intentP_bf, idx3);
  // 5) segment softmax + sparse aggregation into intents
  k_zero_u32<<<(NC_ + 255) / 256, 256, 0, stream>>>(m1u, NC_);
  k_zero_u32<<<(NC_ + 255) / 256, 256, 0, stream>>>((unsigned*)den1, NC_);
  k_zero_u32<<<(NC_ * D_ + 255) / 256, 256, 0, stream>>>((unsigned*)intent_new, NC_ * D_);
  k_edge_e1<<<(E_ + 3) / 4, 128, 0, stream>>>(item_w, intent_w, la, idx3, e1, m1u, E_);
  k_edge_ex<<<(E_ + 255) / 256, 256, 0, stream>>>(e1, idx3, m1u, ex1, den1, E_);
  k_edge_scatter<<<(E_ + 3) / 4, 128, 0, stream>>>(item_w, idx3, ex1, den1, intent_new, E_);
  // 6) per-item softmax over 3 intents -> agg
  k_item_nei<<<(NI_ + 3) / 4, 128, 0, stream>>>(item_w, intent_new, lb, idx3, agg, NI_);
  // 7) session means
  k_hs<<<(B_ + 3) / 4, 128, 0, stream>>>(item_w, inputs, masks, hs);
  // 8) per-session dense attention
  k_local<<<B_, 256, 0, stream>>>(item_w, agg, hs, items, Amat, la1, la2, lb1, lb2, outBND);
  // 9) readout
  k_sessions<<<B_, 256, 0, stream>>>(outBND, alias, masks, pos_emb,
                                     sessions, hir_bf, hsr, hsr_bf);
  k_gemm_bt_bf16<false><<<dim3(B_ * L_ / 16, D_ / 16), 32, 0, stream>>>(
      hir_bf, W1T_bf, q1, B_ * L_, D_, D_, 1.f);
  k_gemm_bt_bf16<false><<<dim3(B_ / 16, D_ / 16), 32, 0, stream>>>(
      hsr_bf, W2T_bf, q2, B_, D_, D_, 1.f);
  k_al<<<(B_ * L_ + 3) / 4, 128, 0, stream>>>(q1, q2, b1, b2, w3, al);
  k_hf<<<B_, 128, 0, stream>>>(sessions, al, masks, pos_emb, hf, hf_bf);
  // 10) scores = hf @ item_w[1:].T  (WMMA, writes d_out [512 x 99999])
  k_gemm_bt_bf16<false><<<dim3(B_ / 16, (NI_ - 1 + 15) / 16), 32, 0, stream>>>(
      hf_bf, item_w_bf + D_, (float*)d_out, B_, NI_ - 1, D_, 1.f);
}
